// QuantizeNIN_24816321036974
// MI455X (gfx1250) — matrix operand order repaired and checked by reading, not compiled, not run on hardware
//
#include <hip/hip_runtime.h>
#include <hip/hip_bf16.h>

typedef _Float16 half8  __attribute__((ext_vector_type(8)));
typedef _Float16 half16 __attribute__((ext_vector_type(16)));
typedef float    float8 __attribute__((ext_vector_type(8)));
typedef int      v4i    __attribute__((vector_size(16)));

#define NB 512   // batch size (512*3*32*32 input)

// CDNA5 async global->LDS path (guarded: falls back to reg staging if the
// toolchain does not declare the gfx1250 async builtins)
#if defined(__HIP_DEVICE_COMPILE__)
#if __has_builtin(__builtin_amdgcn_global_load_async_to_lds_b128) && \
    __has_builtin(__builtin_amdgcn_s_wait_asynccnt)
#define USE_ASYNC_LDS 1
#endif
#endif

#ifdef USE_ASYNC_LDS
typedef __attribute__((address_space(1))) v4i as1_v4i;   // global b128 payload
typedef __attribute__((address_space(3))) v4i as3_v4i;   // LDS b128 payload
#endif

__device__ __forceinline__ half8 h8zero() {
  half8 v;
#pragma unroll
  for (int i = 0; i < 8; ++i) v[i] = (_Float16)0.0f;
  return v;
}

// ---------------------------------------------------------------------------
// prep: x f32 NCHW [512,3,32,32] -> f16 NHWC padded to 8 channels [512,32,32,8]
// ---------------------------------------------------------------------------
__global__ void prep_x_kernel(const float* __restrict__ x, _Float16* __restrict__ xp) {
  int idx = blockIdx.x * 256 + threadIdx.x;
  const int total = NB * 32 * 32;
  if (idx >= total) return;
  int b = idx >> 10;          // / 1024
  int r = idx & 1023;         // y*32 + x
  half8 v = h8zero();
  v[0] = (_Float16)x[((size_t)b * 3 + 0) * 1024 + r];
  v[1] = (_Float16)x[((size_t)b * 3 + 1) * 1024 + r];
  v[2] = (_Float16)x[((size_t)b * 3 + 2) * 1024 + r];
  *(half8*)(xp + (size_t)idx * 8) = v;
}

// ---------------------------------------------------------------------------
// prep: w f32 OIHW [CO][CI][KH*KW] -> f16 [CO][KH*KW][CIpad]  (K contiguous)
// ---------------------------------------------------------------------------
__global__ void prep_w_kernel(const float* __restrict__ w, _Float16* __restrict__ wt,
                              int CO, int CI, int KHW, int CIp) {
  int idx = blockIdx.x * 256 + threadIdx.x;
  int total = CO * KHW * CIp;
  if (idx >= total) return;
  int ci   = idx % CIp;
  int t    = idx / CIp;
  int cell = t % KHW;
  int co   = t / KHW;
  float v = (ci < CI) ? w[((size_t)co * CI + ci) * KHW + cell] : 0.0f;
  wt[idx] = (_Float16)v;
}

// ---------------------------------------------------------------------------
// Implicit-GEMM conv (stride 1, square kernel, S x S image) + bias + ReLU.
//   in : f16 NHWC [NB, S, S, CI]   (CI % 8 == 0)
//   wt : f16      [CO][KH*KW*CI]   (K contiguous, K % 8 == 0)
//   out: f16 NHWC [NB, S, S, CO]
// GEMM: M = CO, N = NB*S*S, K = KH*KH*CI
// Block tile: 64(M) x 256(N) x 32(K). 8 waves; each wave: 64x32 = 4x2 WMMA frags.
// All shape parameters are compile-time -> no runtime div/mod on the hot path.
// ---------------------------------------------------------------------------
template<int CI, int CO, int KH, int PAD, int S>
__global__ __launch_bounds__(256) void conv_wmma_kernel(
    const _Float16* __restrict__ in, const _Float16* __restrict__ wt,
    const float* __restrict__ bias, _Float16* __restrict__ out)
{
  constexpr int KW = KH;
  constexpr int K  = KH * KW * CI;
  constexpr int KT = (K + 31) >> 5;
  constexpr int HW = S * S;         // power of two for all layers

  const int tid  = threadIdx.x;
  const int wave = tid >> 5;
  const int lane = tid & 31;
  const int g    = lane >> 4;    // half-wave group (0/1)
  const int lr   = lane & 15;

  const int n1 = blockIdx.x * 256;  // pixel tile base
  const int m0 = blockIdx.y * 64;   // out-channel tile base

  // LDS tiles, rows padded 32 -> 40 halfs (80B): bank-conflict-free frag loads,
  // 16B alignment kept for b128 ds / async ops
  __shared__ _Float16 As[2][64][40];   // [buf][m][k]
  __shared__ _Float16 Bs[2][256][40];  // [buf][n][k]

  // this thread's B row: one output pixel, all 32 K values per k-step
  const int npix = n1 + tid;
  const int b    = npix / HW;       // HW = 1024/256/64 -> shifts
  const int rr   = npix % HW;
  const int oy   = rr / S;
  const int ox   = rr % S;

  // this thread's A chunk: row am, 8 halfs at k-offset ak
  const int am = tid >> 2;
  const int ak = (tid & 3) << 3;

  float8 acc[4][2];
#pragma unroll
  for (int mt = 0; mt < 4; ++mt)
#pragma unroll
    for (int u = 0; u < 2; ++u)
#pragma unroll
      for (int r = 0; r < 8; ++r) acc[mt][u][r] = 0.0f;

  // ---- per-chunk source address/validity (compile-time folded for 1x1) ----
  auto b_src = [&](int kk, bool& valid) -> const _Float16* {
    int cell = kk / CI;             // constant divisor
    int ci   = kk % CI;
    int kh   = cell / KW;
    int kw   = cell % KW;
    int iy   = oy + kh - PAD;
    int ix   = ox + kw - PAD;
    valid = (kk < K) && (iy >= 0) && (iy < S) && (ix >= 0) && (ix < S);
    return in + (((size_t)b * S + iy) * S + ix) * CI + ci;
  };

  auto compute = [&](int bi) {
    half16 af[4], bf[2];
    // A frag (16x32 f16): lane holds row lr; K chunks at 8g and 16+8g
#pragma unroll
    for (int mt = 0; mt < 4; ++mt) {
      half8 lo = *(const half8*)&As[bi][mt * 16 + lr][g * 8];
      half8 hi = *(const half8*)&As[bi][mt * 16 + lr][16 + g * 8];
#pragma unroll
      for (int i = 0; i < 8; ++i) { af[mt][i] = lo[i]; af[mt][i + 8] = hi[i]; }
    }
    // B frag (32x16 f16): lane holds column lr; K chunks at 8g and 16+8g
#pragma unroll
    for (int u = 0; u < 2; ++u) {
      int row = wave * 32 + u * 16 + lr;
      half8 lo = *(const half8*)&Bs[bi][row][g * 8];
      half8 hi = *(const half8*)&Bs[bi][row][16 + g * 8];
#pragma unroll
      for (int i = 0; i < 8; ++i) { bf[u][i] = lo[i]; bf[u][i + 8] = hi[i]; }
    }
#pragma unroll
    for (int mt = 0; mt < 4; ++mt)
#pragma unroll
      for (int u = 0; u < 2; ++u)
        acc[mt][u] = __builtin_amdgcn_wmma_f32_16x16x32_f16(
            false, af[mt], false, bf[u], (short)0, acc[mt][u], false, false);
  };

#ifdef USE_ASYNC_LDS
  // -------- CDNA5 path: GLOBAL_LOAD_ASYNC_TO_LDS_B128, one barrier / step ----
  auto issue_tile = [&](int kt, int bi) {
    const int k0 = kt << 5;
    {   // A chunk (weights)
      int gk = k0 + ak;
      int co = m0 + am;
      _Float16* lp = &As[bi][am][ak];
      if (co < CO && gk < K) {
        __builtin_amdgcn_global_load_async_to_lds_b128(
            (as1_v4i*)(wt + (size_t)co * K + gk), (as3_v4i*)lp, 0, 0);
      } else {
        *(half8*)lp = h8zero();           // ds_store zero-fill
      }
    }
#pragma unroll
    for (int c = 0; c < 4; ++c) {         // B chunks (activations)
      int kk = k0 + c * 8;
      bool valid;
      const _Float16* gp = b_src(kk, valid);
      _Float16* lp = &Bs[bi][tid][c * 8];
      if (valid) {
        __builtin_amdgcn_global_load_async_to_lds_b128(
            (as1_v4i*)gp, (as3_v4i*)lp, 0, 0);
      } else {
        *(half8*)lp = h8zero();
      }
    }
  };

  issue_tile(0, 0);
  __builtin_amdgcn_s_wait_asynccnt(0);
  __syncthreads();
  for (int kt = 0; kt < KT; ++kt) {
    int cur = kt & 1;
    if (kt + 1 < KT) issue_tile(kt + 1, cur ^ 1);  // async engine fills other buf
    compute(cur);                                   // WMMA on current buf
    if (kt + 1 < KT) __builtin_amdgcn_s_wait_asynccnt(0);
    __syncthreads();
  }
#else
  // -------- fallback: global -> VGPR -> ds_store, double-buffered ----------
  half8 aR;
  half8 bR[4];
  auto load_global = [&](int kt) {
    const int k0 = kt << 5;
    {
      int gk = k0 + ak;
      int co = m0 + am;
      aR = h8zero();
      if (co < CO && gk < K)
        aR = *(const half8*)(wt + (size_t)co * K + gk);
    }
#pragma unroll
    for (int c = 0; c < 4; ++c) {
      int kk = k0 + c * 8;
      bool valid;
      const _Float16* gp = b_src(kk, valid);
      bR[c] = valid ? *(const half8*)gp : h8zero();
    }
  };
  auto store_lds = [&](int bi) {
    *(half8*)&As[bi][am][ak] = aR;
#pragma unroll
    for (int c = 0; c < 4; ++c)
      *(half8*)&Bs[bi][tid][c * 8] = bR[c];
  };

  load_global(0);
  store_lds(0);
  __syncthreads();
  for (int kt = 0; kt < KT; ++kt) {
    int cur = kt & 1;
    if (kt + 1 < KT) load_global(kt + 1);
    compute(cur);
    __syncthreads();
    if (kt + 1 < KT) {
      store_lds(cur ^ 1);
      __syncthreads();
    }
  }
#endif

  // epilogue: D layout -> lane holds (m = r + 8g, n = lr); fused bias + ReLU
#pragma unroll
  for (int mt = 0; mt < 4; ++mt) {
#pragma unroll
    for (int u = 0; u < 2; ++u) {
      int co0 = m0 + mt * 16 + g * 8;
      int np  = n1 + wave * 32 + u * 16 + lr;
      _Float16* op = out + (size_t)np * CO + co0;
      if (((CO & 7) == 0) && (co0 + 8 <= CO)) {   // folds at compile time
        half8 hv;
#pragma unroll
        for (int r = 0; r < 8; ++r)
          hv[r] = (_Float16)fmaxf(acc[mt][u][r] + bias[co0 + r], 0.0f);
        *(half8*)op = hv;
      } else {
#pragma unroll
        for (int r = 0; r < 8; ++r)
          if (co0 + r < CO)
            op[r] = (_Float16)fmaxf(acc[mt][u][r] + bias[co0 + r], 0.0f);
      }
    }
  }
}

// ---------------------------------------------------------------------------
// pool 3x3 stride 2 pad 1 + sign() quantizer.  AVG=false: maxpool(-inf pad);
// AVG=true: avgpool with count_include_pad (/9).  All dims compile-time.
// ---------------------------------------------------------------------------
template<int IH, int OH, int C, bool AVG>
__global__ void pool_sign_kernel(const _Float16* __restrict__ in, _Float16* __restrict__ out) {
  int idx = blockIdx.x * 256 + threadIdx.x;
  constexpr int total = NB * OH * OH * C;
  if (idx >= total) return;
  int c = idx % C; int t = idx / C;
  int ox = t % OH; t /= OH;
  int oy = t % OH; int b = t / OH;
  float m = AVG ? 0.0f : -1e30f;
#pragma unroll
  for (int dy = 0; dy < 3; ++dy) {
#pragma unroll
    for (int dx = 0; dx < 3; ++dx) {
      int iy = oy * 2 - 1 + dy;
      int ix = ox * 2 - 1 + dx;
      if (iy >= 0 && iy < IH && ix >= 0 && ix < IH) {
        float v = (float)in[(((size_t)b * IH + iy) * IH + ix) * C + c];
        m = AVG ? (m + v) : fmaxf(m, v);
      }
    }
  }
  if (AVG) m *= (1.0f / 9.0f);
  float s = (m > 0.0f) ? 1.0f : ((m < 0.0f) ? -1.0f : 0.0f);
  out[idx] = (_Float16)s;
}

// ---------------------------------------------------------------------------
// final: avgpool 8x8 on [512,8,8,10] -> f32 [512,10]
// ---------------------------------------------------------------------------
__global__ void final_avg_kernel(const _Float16* __restrict__ in, float* __restrict__ out) {
  int idx = blockIdx.x * 256 + threadIdx.x;
  if (idx >= NB * 10) return;
  int n = idx / 10, c = idx % 10;
  float s = 0.0f;
  for (int p = 0; p < 64; ++p)
    s += (float)in[((size_t)n * 64 + p) * 10 + c];
  out[idx] = s * (1.0f / 64.0f);
}

// ---------------------------------------------------------------------------
template<int CI, int CO, int KH, int PAD, int S>
static void launch_conv(const _Float16* in, const _Float16* wt, const float* bias,
                        _Float16* out, hipStream_t stream) {
  constexpr int N = NB * S * S;               // multiple of 256 for all layers
  dim3 grid(N / 256, (CO + 63) / 64);
  conv_wmma_kernel<CI, CO, KH, PAD, S><<<grid, 256, 0, stream>>>(in, wt, bias, out);
}

extern "C" void kernel_launch(void* const* d_in, const int* in_sizes, int n_in,
                              void* d_out, int out_size, void* d_ws, size_t ws_size,
                              hipStream_t stream) {
  (void)in_sizes; (void)n_in; (void)out_size; (void)ws_size;
  const float* x = (const float*)d_in[0];
  const float* Wf[9]; const float* Bf[9];
  for (int i = 0; i < 9; ++i) { Wf[i] = (const float*)d_in[1 + 2 * i]; Bf[i] = (const float*)d_in[2 + 2 * i]; }

  // layer tables (CIp = padded input channels, multiple of 8)
  const int CIp[9]    = {8, 192, 160, 96, 192, 192, 192, 192, 192};
  const int CIreal[9] = {3, 192, 160, 96, 192, 192, 192, 192, 192};
  const int COt[9]    = {192, 160, 96, 192, 192, 192, 192, 192, 10};
  const int KHt[9]    = {5, 1, 1, 5, 1, 1, 3, 1, 1};

  // workspace carve-out (all offsets 256B aligned)
  char* ws = (char*)d_ws;
  size_t off = 0;
  auto alloc = [&](size_t halfs) -> _Float16* {
    _Float16* p = (_Float16*)(ws + off);
    off = (off + halfs * 2 + 255) & ~(size_t)255;
    return p;
  };
  _Float16* xpad = alloc((size_t)NB * 32 * 32 * 8);     //   8.4 MB
  _Float16* bufA = alloc((size_t)NB * 32 * 32 * 192);   // 201.3 MB
  _Float16* bufB = alloc((size_t)NB * 32 * 32 * 160);   // 167.8 MB
  _Float16* wt[9];
  for (int i = 0; i < 9; ++i)
    wt[i] = alloc((size_t)COt[i] * KHt[i] * KHt[i] * CIp[i]);

  // ---- prep ----
  {
    int total = NB * 32 * 32;
    prep_x_kernel<<<(total + 255) / 256, 256, 0, stream>>>(x, xpad);
  }
  for (int i = 0; i < 9; ++i) {
    int khw = KHt[i] * KHt[i];
    int total = COt[i] * khw * CIp[i];
    prep_w_kernel<<<(total + 255) / 256, 256, 0, stream>>>(Wf[i], wt[i], COt[i], CIreal[i], khw, CIp[i]);
  }

  // ---- network ----
  launch_conv<8,   192, 5, 2, 32>(xpad, wt[0], Bf[0], bufA, stream);  // [512,32,32,192]
  launch_conv<192, 160, 1, 0, 32>(bufA, wt[1], Bf[1], bufB, stream);  // [512,32,32,160]
  launch_conv<160,  96, 1, 0, 32>(bufB, wt[2], Bf[2], bufA, stream);  // [512,32,32,96]
  {
    constexpr int total = NB * 16 * 16 * 96;
    pool_sign_kernel<32, 16, 96, false><<<(total + 255) / 256, 256, 0, stream>>>(bufA, bufB);
  }
  launch_conv<96,  192, 5, 2, 16>(bufB, wt[3], Bf[3], bufA, stream);  // [512,16,16,192]
  launch_conv<192, 192, 1, 0, 16>(bufA, wt[4], Bf[4], bufB, stream);
  launch_conv<192, 192, 1, 0, 16>(bufB, wt[5], Bf[5], bufA, stream);
  {
    constexpr int total = NB * 8 * 8 * 192;
    pool_sign_kernel<16, 8, 192, true><<<(total + 255) / 256, 256, 0, stream>>>(bufA, bufB);
  }
  launch_conv<192, 192, 3, 1, 8>(bufB, wt[6], Bf[6], bufA, stream);   // [512,8,8,192]
  launch_conv<192, 192, 1, 0, 8>(bufA, wt[7], Bf[7], bufB, stream);
  launch_conv<192,  10, 1, 0, 8>(bufB, wt[8], Bf[8], bufA, stream);   // [512,8,8,10]
  {
    constexpr int total = NB * 10;
    final_avg_kernel<<<(total + 255) / 256, 256, 0, stream>>>(bufA, (float*)d_out);
  }
}
